// ResidualAttentionBlock_9002251453005
// MI455X (gfx1250) — matrix operand order repaired and checked by reading, not compiled, not run on hardware
//
#include <hip/hip_runtime.h>
#include <hip/hip_bf16.h>
#include <math.h>

// ---------------- problem constants ----------------
constexpr int kSeq   = 1025;             // STRIDE + 1
constexpr int kSeqP  = 1040;             // seq padded to mult of 16 (rows 2080B, 16B aligned)
constexpr int kBatch = 4;
constexpr int kC     = 1024;             // D_MODEL
constexpr int kHeads = 16;
constexpr int kDh    = 64;
constexpr int kTok   = kSeq * kBatch;    // 4100 token rows, layout [N,B,C] flattened
constexpr int kMT    = (kSeq + 15) / 16; // 65 seq tiles

typedef __bf16 bf16_t;
typedef __attribute__((ext_vector_type(16))) __bf16 v16bf;
typedef __attribute__((ext_vector_type(8)))  __bf16 v8bf;
typedef __attribute__((ext_vector_type(8)))  float  v8f;

static __device__ __forceinline__ v16bf cat8(v8bf lo, v8bf hi) {
  return __builtin_shufflevector(lo, hi, 0,1,2,3,4,5,6,7,8,9,10,11,12,13,14,15);
}
static __device__ __forceinline__ v8f wmma_bf16(v16bf a, v16bf b, v8f c) {
  // (neg_a, A, neg_b, B, c_mod, C, reuse_a, reuse_b) -> v_wmma_f32_16x16x32_bf16
  return __builtin_amdgcn_wmma_f32_16x16x32_bf16(false, a, false, b, (short)0, c, false, false);
}

// CDNA5 async global->LDS copy (ASYNCcnt-tracked); lds_off = LDS byte offset
// (low 32 bits of a generic LDS pointer).
static __device__ __forceinline__ void async_ld16(unsigned lds_off, const void* gaddr) {
  asm volatile("global_load_async_to_lds_b128 %0, %1, off"
               :: "v"(lds_off), "v"(gaddr) : "memory");
}
static __device__ __forceinline__ void wait_async0() {
  asm volatile("s_wait_asynccnt 0x0" ::: "memory");
}

// ---------------- f32 -> bf16 weight conversion ----------------
__global__ void cvt_f32_bf16(const float* __restrict__ s, bf16_t* __restrict__ d, int n) {
  int i = blockIdx.x * blockDim.x + threadIdx.x;
  int stride = gridDim.x * blockDim.x;
  for (; i < n; i += stride) d[i] = (bf16_t)s[i];
}

// ---------------- LayerNorm: one 256-thread block per 1024-wide row ----------------
__global__ void layernorm_bf16(const float* __restrict__ x, const float* __restrict__ g,
                               const float* __restrict__ b, bf16_t* __restrict__ y) {
  __shared__ float s0[256], s1[256];
  const int row = blockIdx.x, tid = threadIdx.x;
  const float* xr = x + (size_t)row * kC;
  float v[4], s = 0.f, q = 0.f;
#pragma unroll
  for (int i = 0; i < 4; i++) { v[i] = xr[tid + i * 256]; s += v[i]; q += v[i] * v[i]; }
  s0[tid] = s; s1[tid] = q; __syncthreads();
  for (int o = 128; o > 0; o >>= 1) {
    if (tid < o) { s0[tid] += s0[tid + o]; s1[tid] += s1[tid + o]; }
    __syncthreads();
  }
  const float mean = s0[0] * (1.f / kC);
  const float var  = s1[0] * (1.f / kC) - mean * mean;
  const float rstd = rsqrtf(var + 1e-5f);
  bf16_t* yr = y + (size_t)row * kC;
#pragma unroll
  for (int i = 0; i < 4; i++) {
    int c = tid + i * 256;
    yr[c] = (bf16_t)((v[i] - mean) * rstd * g[c] + b[c]);
  }
}

// ---------------- generic tiled WMMA GEMM: D = A[M,K] * W[N,K]^T (+bias)(+gelu)(+res) --------
// block tile 64(M) x 128(N), BK=32, 256 threads = 8 waves, wave tile 32x32 (2x2 WMMA).
// Double-buffered LDS staging through GLOBAL_LOAD_ASYNC_TO_LDS_B128 (ASYNCcnt).
template<int ACT, bool OUT_BF16, bool HAS_BIAS, bool HAS_RES>
__global__ void gemm_wmma(const bf16_t* __restrict__ A, const bf16_t* __restrict__ W,
                          const float* __restrict__ bias, const float* __restrict__ res,
                          float* __restrict__ Df, bf16_t* __restrict__ Db,
                          int M, int N, int K) {
  __shared__ bf16_t As[2][64][40];    // padded stride 40 elems = 80B (16B aligned rows)
  __shared__ bf16_t Bs[2][128][40];
  const int tid = threadIdx.x;
  const int lane = tid & 31, wid = tid >> 5;
  const int hf = lane >> 4, lc = lane & 15;
  const int wm = wid >> 2, wn = wid & 3;         // 2 x 4 wave grid
  const int m0 = blockIdx.y * 64, n0 = blockIdx.x * 128;
  const int lr  = tid >> 2;                      // 0..63 staging row
  const int lcg = (tid & 3) * 8;                 // 0,8,16,24 staging col group

  int ga = m0 + lr; if (ga > M - 1) ga = M - 1;  // clamp M edge (store guarded)
  const bf16_t* gA  = A + (size_t)ga * K + lcg;
  const bf16_t* gB0 = W + (size_t)(n0 + lr) * K + lcg;
  const bf16_t* gB1 = W + (size_t)(n0 + lr + 64) * K + lcg;
  const unsigned ldsA  = (unsigned)(size_t)&As[0][lr][lcg];
  const unsigned ldsB0 = (unsigned)(size_t)&Bs[0][lr][lcg];
  const unsigned ldsB1 = (unsigned)(size_t)&Bs[0][lr + 64][lcg];
  const unsigned bufA = sizeof(As[0]);           // 5120 B
  const unsigned bufB = sizeof(Bs[0]);           // 10240 B

  v8f acc[2][2] = {};
  // prologue: stage k-step 0 into buffer 0
  async_ld16(ldsA,  gA);
  async_ld16(ldsB0, gB0);
  async_ld16(ldsB1, gB1);
  const int nk = K >> 5;
  for (int ki = 0; ki < nk; ki++) {
    const int cur = ki & 1, nxt = cur ^ 1;
    wait_async0();          // this wave's async writes into buf[cur] landed
    __syncthreads();        // everyone's landed; prior reads of buf[nxt] finished
    if (ki + 1 < nk) {      // stage next k-step into the other buffer, no wait
      const int k1 = (ki + 1) << 5;
      async_ld16(ldsA  + nxt * bufA, gA  + k1);
      async_ld16(ldsB0 + nxt * bufB, gB0 + k1);
      async_ld16(ldsB1 + nxt * bufB, gB1 + k1);
    }
    v16bf afrag[2], bfrag[2];
#pragma unroll
    for (int t = 0; t < 2; t++) {
      const int ar = wm * 32 + t * 16 + lc;
      afrag[t] = cat8(*(const v8bf*)&As[cur][ar][hf * 8],
                      *(const v8bf*)&As[cur][ar][16 + hf * 8]);
      const int bc = wn * 32 + t * 16 + lc;
      bfrag[t] = cat8(*(const v8bf*)&Bs[cur][bc][hf * 8],
                      *(const v8bf*)&Bs[cur][bc][16 + hf * 8]);
    }
#pragma unroll
    for (int i = 0; i < 2; i++)
#pragma unroll
      for (int j = 0; j < 2; j++)
        acc[i][j] = wmma_bf16(afrag[i], bfrag[j], acc[i][j]);
  }
  // epilogue: bias -> act -> residual -> store
#pragma unroll
  for (int i = 0; i < 2; i++) {
#pragma unroll
    for (int j = 0; j < 2; j++) {
      const int row0 = m0 + wm * 32 + i * 16 + hf * 8;
      const int col  = n0 + wn * 32 + j * 16 + lc;
#pragma unroll
      for (int e = 0; e < 8; e++) {
        const int row = row0 + e;
        if (row < M) {
          float v = acc[i][j][e];
          if (HAS_BIAS) v += bias[col];
          if (ACT == 1) v = v / (1.f + __expf(-1.702f * v));   // quick_gelu
          if (HAS_RES)  v += res[(size_t)row * N + col];
          if (OUT_BF16) Db[(size_t)row * N + col] = (bf16_t)v;
          else          Df[(size_t)row * N + col] = v;
        }
      }
    }
  }
}

// ---------------- attention scores: one wave per 16x16 tile, causal skip ----------------
__global__ void attn_scores(const bf16_t* __restrict__ qkv, float* __restrict__ attn) {
  const int t = blockIdx.x;
  const int mt = t / kMT, nt = t % kMT;
  if (nt > mt) return;                              // fully masked tile
  const int bh = blockIdx.y;
  const int b = bh >> 4, h = bh & 15;
  const int lane = threadIdx.x & 31;
  const int hf = lane >> 4, lc = lane & 15;
  int n = mt * 16 + lc; if (n > kSeq - 1) n = kSeq - 1;
  int m = nt * 16 + lc; if (m > kSeq - 1) m = kSeq - 1;
  const bf16_t* qrow = qkv + (size_t)(n * kBatch + b) * (3 * kC) + h * kDh;          // q
  const bf16_t* krow = qkv + (size_t)(m * kBatch + b) * (3 * kC) + kC + h * kDh;     // k
  v8f acc = {};
#pragma unroll
  for (int k0 = 0; k0 < kDh; k0 += 32) {
    v16bf a  = cat8(*(const v8bf*)(qrow + k0 + hf * 8),
                    *(const v8bf*)(qrow + k0 + 16 + hf * 8));
    v16bf bb = cat8(*(const v8bf*)(krow + k0 + hf * 8),
                    *(const v8bf*)(krow + k0 + 16 + hf * 8));
    acc = wmma_bf16(a, bb, acc);
  }
  const float scale = 0.125f;                       // Dh^-0.5 = 1/8
  const int rn0 = mt * 16 + hf * 8;
  const int cm  = nt * 16 + lc;
  if (cm < kSeq) {
    float* pout = attn + ((size_t)bh * kSeq + rn0) * kSeq + cm;
#pragma unroll
    for (int e = 0; e < 8; e++)
      if (rn0 + e < kSeq) pout[(size_t)e * kSeq] = acc[e] * scale;
  }
}

// ---------------- causal softmax: f32 probs to d_out + padded bf16 copy to ws -------------
__global__ void softmax_causal(float* __restrict__ attn, bf16_t* __restrict__ attnp) {
  __shared__ float red[256];
  const int r = blockIdx.x;                 // bh*SEQ + n
  const int n = r % kSeq;
  const int tid = threadIdx.x;
  float* row = attn + (size_t)r * kSeq;
  bf16_t* prow = attnp + (size_t)r * kSeqP;
  const int valid = n + 1;
  float mx = -3.4e38f;
  for (int m = tid; m < valid; m += 256) mx = fmaxf(mx, row[m]);
  red[tid] = mx; __syncthreads();
  for (int o = 128; o > 0; o >>= 1) { if (tid < o) red[tid] = fmaxf(red[tid], red[tid + o]); __syncthreads(); }
  mx = red[0]; __syncthreads();
  float s = 0.f;
  for (int m = tid; m < valid; m += 256) s += __expf(row[m] - mx);
  red[tid] = s; __syncthreads();
  for (int o = 128; o > 0; o >>= 1) { if (tid < o) red[tid] += red[tid + o]; __syncthreads(); }
  const float inv = 1.f / red[0];
  for (int m = tid; m < kSeqP; m += 256) {
    float val = (m < valid) ? __expf(row[m] - mx) * inv : 0.f;
    if (m < kSeq) row[m] = val;             // masked region explicitly zeroed
    prow[m] = (bf16_t)val;                  // padded bf16 copy for WMMA attn@V
  }
}

// ---------------- V transpose: vt[bh][d][m] bf16, m padded to kSeqP -----------------------
__global__ void build_vt(const bf16_t* __restrict__ qkv, bf16_t* __restrict__ vt) {
  __shared__ bf16_t tile[16][72];
  const int mt = blockIdx.x;                // 65
  const int bh = blockIdx.y;                // 64
  const int b = bh >> 4, h = bh & 15;
  const int t = threadIdx.x;                // 256
  {
    const int mi = t >> 4;                  // 0..15
    const int d0 = (t & 15) * 4;            // 0..60
    const int m = mt * 16 + mi;
    bf16_t v0 = (bf16_t)0.f, v1 = v0, v2 = v0, v3 = v0;
    if (m < kSeq) {
      const bf16_t* src = qkv + (size_t)(m * kBatch + b) * (3 * kC) + 2 * kC + h * kDh + d0;
      v0 = src[0]; v1 = src[1]; v2 = src[2]; v3 = src[3];
    }
    tile[mi][d0] = v0; tile[mi][d0 + 1] = v1; tile[mi][d0 + 2] = v2; tile[mi][d0 + 3] = v3;
  }
  __syncthreads();
  {
    const int d   = t >> 2;                 // 0..63
    const int mi0 = (t & 3) * 4;            // 0,4,8,12
    bf16_t* dst = vt + ((size_t)bh * kDh + d) * kSeqP + mt * 16 + mi0;
    dst[0] = tile[mi0][d]; dst[1] = tile[mi0 + 1][d];
    dst[2] = tile[mi0 + 2][d]; dst[3] = tile[mi0 + 3][d];
  }
}

// ---------------- attn @ V: one wave per 16-row tile, all-vector loads --------------------
__global__ void attn_v(const bf16_t* __restrict__ attnp, const bf16_t* __restrict__ vt,
                       bf16_t* __restrict__ ctx) {
  const int mt = blockIdx.x;
  const int bh = blockIdx.y;
  const int b = bh >> 4, h = bh & 15;
  const int lane = threadIdx.x & 31;
  const int hf = lane >> 4, lc = lane & 15;
  int n = mt * 16 + lc; if (n > kSeq - 1) n = kSeq - 1;
  const bf16_t* arow  = attnp + ((size_t)bh * kSeq + n) * kSeqP;
  const bf16_t* vbase = vt + (size_t)bh * kDh * kSeqP;
  v8f acc[4] = {};
  const int kend = mt * 16 + 16;            // causal: only m <= mt*16+15 contribute
  for (int k0 = 0; k0 < kend; k0 += 32) {   // padded rows are zero -> no guards needed
    v16bf a = cat8(*(const v8bf*)(arow + k0 + hf * 8),
                   *(const v8bf*)(arow + k0 + 16 + hf * 8));
#pragma unroll
    for (int d = 0; d < 4; d++) {
      const bf16_t* vr = vbase + (size_t)(d * 16 + lc) * kSeqP;
      v16bf bv = cat8(*(const v8bf*)(vr + k0 + hf * 8),
                      *(const v8bf*)(vr + k0 + 16 + hf * 8));
      acc[d] = wmma_bf16(a, bv, acc[d]);
    }
  }
#pragma unroll
  for (int d = 0; d < 4; d++) {
#pragma unroll
    for (int e = 0; e < 8; e++) {
      const int rn = mt * 16 + hf * 8 + e;
      if (rn < kSeq)
        ctx[(size_t)(rn * kBatch + b) * kC + h * kDh + d * 16 + lc] = (bf16_t)acc[d][e];
    }
  }
}

// ---------------- host-side orchestration ----------------
extern "C" void kernel_launch(void* const* d_in, const int* in_sizes, int n_in,
                              void* d_out, int out_size, void* d_ws, size_t ws_size,
                              hipStream_t stream) {
  (void)in_sizes; (void)n_in; (void)out_size; (void)ws_size;
  const float* x       = (const float*)d_in[0];   // [N,B,C]
  const float* qkv_w   = (const float*)d_in[1];   // [3C,C]
  const float* proj_w  = (const float*)d_in[2];   // [C,C]
  const float* proj_b  = (const float*)d_in[3];
  const float* ln1_g   = (const float*)d_in[4];
  const float* ln1_b   = (const float*)d_in[5];
  const float* ln2_g   = (const float*)d_in[6];
  const float* ln2_b   = (const float*)d_in[7];
  const float* fc_w    = (const float*)d_in[8];   // [4C,C]
  const float* fc_b    = (const float*)d_in[9];
  const float* cproj_w = (const float*)d_in[10];  // [C,4C]
  const float* cproj_b = (const float*)d_in[11];

  float* x_out = (float*)d_out;                        // [N,B,C] final block output
  float* attn  = x_out + (size_t)kSeq * kBatch * kC;   // [B,H,N,N] softmax probs

  // workspace carve-out (~271 MB total)
  char* p = (char*)d_ws;
  auto carve = [&](size_t bytes) { void* r = p; p += (bytes + 255) & ~(size_t)255; return r; };
  bf16_t* w_qkv   = (bf16_t*)carve((size_t)3 * kC * kC * 2);
  bf16_t* w_proj  = (bf16_t*)carve((size_t)kC * kC * 2);
  bf16_t* w_fc    = (bf16_t*)carve((size_t)4 * kC * kC * 2);
  bf16_t* w_cproj = (bf16_t*)carve((size_t)4 * kC * kC * 2);
  bf16_t* ln1o    = (bf16_t*)carve((size_t)kTok * kC * 2);
  bf16_t* qkvb    = (bf16_t*)carve((size_t)kTok * 3 * kC * 2);
  bf16_t* ctx     = (bf16_t*)carve((size_t)kTok * kC * 2);
  float*  x1      = (float*) carve((size_t)kTok * kC * 4);
  bf16_t* ln2o    = (bf16_t*)carve((size_t)kTok * kC * 2);
  bf16_t* hbuf    = (bf16_t*)carve((size_t)kTok * 4 * kC * 2);
  bf16_t* attnp   = (bf16_t*)carve((size_t)kBatch * kHeads * kSeq * kSeqP * 2); // 136 MB
  bf16_t* vtb     = (bf16_t*)carve((size_t)kBatch * kHeads * kDh * kSeqP * 2);  // 8.5 MB

  // 0) bf16 weight staging (L2-resident thereafter: ~25 MB << 192 MB L2)
  cvt_f32_bf16<<<2048, 256, 0, stream>>>(qkv_w,   w_qkv,   3 * kC * kC);
  cvt_f32_bf16<<<2048, 256, 0, stream>>>(proj_w,  w_proj,  kC * kC);
  cvt_f32_bf16<<<2048, 256, 0, stream>>>(fc_w,    w_fc,    4 * kC * kC);
  cvt_f32_bf16<<<2048, 256, 0, stream>>>(cproj_w, w_cproj, 4 * kC * kC);

  const dim3 blk(256);
  const int mBlocks = (kTok + 63) / 64;   // 65

  // 1) ln1(x) -> bf16
  layernorm_bf16<<<kTok, blk, 0, stream>>>(x, ln1_g, ln1_b, ln1o);
  // 2) qkv = ln1 @ qkv_w^T  (no bias), bf16 out
  gemm_wmma<0, true, false, false><<<dim3(3 * kC / 128, mBlocks), blk, 0, stream>>>(
      ln1o, w_qkv, nullptr, nullptr, nullptr, qkvb, kTok, 3 * kC, kC);
  // 3) scores -> attn region of d_out (raw logits, causal tiles only)
  attn_scores<<<dim3(kMT * kMT, kBatch * kHeads), dim3(32), 0, stream>>>(qkvb, attn);
  // 3b) vt[bh][d][m] transpose of V (padded, zero-filled tail)
  build_vt<<<dim3(kMT, kBatch * kHeads), blk, 0, stream>>>(qkvb, vtb);
  // 4) causal softmax in place + padded bf16 copy
  softmax_causal<<<kBatch * kHeads * kSeq, blk, 0, stream>>>(attn, attnp);
  // 5) ctx = attn @ V, bf16 out (vectorized fragments from attnp/vt)
  attn_v<<<dim3(kMT, kBatch * kHeads), dim3(32), 0, stream>>>(attnp, vtb, ctx);
  // 6) x1 = ctx @ proj_w^T + proj_b + x   (f32)
  gemm_wmma<0, false, true, true><<<dim3(kC / 128, mBlocks), blk, 0, stream>>>(
      ctx, w_proj, proj_b, x, x1, nullptr, kTok, kC, kC);
  // 7) ln2(x1) -> bf16
  layernorm_bf16<<<kTok, blk, 0, stream>>>(x1, ln2_g, ln2_b, ln2o);
  // 8) h = quick_gelu(ln2 @ fc_w^T + fc_b), bf16 out
  gemm_wmma<1, true, true, false><<<dim3(4 * kC / 128, mBlocks), blk, 0, stream>>>(
      ln2o, w_fc, fc_b, nullptr, nullptr, hbuf, kTok, 4 * kC, kC);
  // 9) x_out = h @ cproj_w^T + cproj_b + x1   (f32, straight into d_out)
  gemm_wmma<0, false, true, true><<<dim3(kC / 128, mBlocks), blk, 0, stream>>>(
      hbuf, w_cproj, cproj_b, x1, x_out, nullptr, kTok, kC, 4 * kC);
}
